// MultiHeadAttention_32023276159342
// MI455X (gfx1250) — matrix operand order repaired and checked by reading, not compiled, not run on hardware
//
#include <hip/hip_runtime.h>
#include <hip/hip_bf16.h>

typedef __attribute__((ext_vector_type(16))) _Float16 v16h;
typedef __attribute__((ext_vector_type(8)))  float    v8f;
typedef __attribute__((ext_vector_type(4)))  unsigned int u32x4;
typedef __attribute__((ext_vector_type(8)))  int      i32x8;
typedef __attribute__((ext_vector_type(4)))  int      i32x4;
typedef _Float16 half_t;

#define S_LEN   2048
#define DMODEL  1024
#define NHEAD   16
#define DK      64
#define BATCH   2
#define MROWS   (BATCH * S_LEN)   // 4096

#if defined(__has_builtin)
#if __has_builtin(__builtin_amdgcn_tensor_load_to_lds) && __has_builtin(__builtin_amdgcn_s_wait_tensorcnt)
#define HAVE_TDM 1
#endif
#endif
#ifndef HAVE_TDM
#define HAVE_TDM 0
#endif

#if HAVE_TDM
// ---------------------------------------------------------------------------
// Tensor Data Mover: 2D tile load Global -> LDS.
// D# built per CDNA5 ISA (CDNA5 ISA):
//   group0: count=1 | lds_addr | global_addr | type=2
//   group1: data_size=2B, tile_dim0/1, tensor_dim0/1, tensor_dim0_stride
// All operands here are wave-uniform (block indices / loop counters).
// ---------------------------------------------------------------------------
__device__ __forceinline__ void tdm_load_2d(unsigned lds_addr,
                                            const void* gptr,
                                            unsigned tile_w, unsigned tile_h,
                                            unsigned tensor_w, unsigned tensor_h,
                                            unsigned long long row_stride)
{
    unsigned long long ga = (unsigned long long)(uintptr_t)gptr;
    u32x4 g0;
    g0[0] = 1u;                                       // count=1, no gather
    g0[1] = lds_addr;                                 // LDS byte address
    g0[2] = (unsigned)(ga & 0xffffffffull);           // global_addr[31:0]
    g0[3] = (unsigned)((ga >> 32) & 0x01ffffffull)    // global_addr[56:32]
          | (2u << 30);                               // type = 2 ("image")
    i32x8 g1;
    g1[0] = (int)(1u << 16);                          // wg_mask=0, data_size=1 (2B)
    g1[1] = (int)((tensor_w & 0xffffu) << 16);        // tensor_dim0[15:0]
    g1[2] = (int)((tensor_w >> 16) | ((tensor_h & 0xffffu) << 16));
    g1[3] = (int)((tensor_h >> 16) | ((tile_w & 0xffffu) << 16));
    g1[4] = (int)(tile_h & 0xffffu);                  // tile_dim1; tile_dim2=0
    g1[5] = (int)(unsigned)(row_stride & 0xffffffffull);   // dim0_stride[31:0]
    g1[6] = (int)(unsigned)((row_stride >> 32) & 0xffffull);
    g1[7] = 0;
    i32x4 z4 = {0, 0, 0, 0};
#if __clang_major__ >= 23
    i32x8 z8 = {0, 0, 0, 0, 0, 0, 0, 0};
    __builtin_amdgcn_tensor_load_to_lds(g0, g1, z4, z4, z8, 0);
#else
    __builtin_amdgcn_tensor_load_to_lds(g0, g1, z4, z4, 0);
#endif
}
__device__ __forceinline__ unsigned lds_off(const void* p) {
    // flat LDS address: low 32 bits are the LDS byte offset
    return (unsigned)(unsigned long long)(uintptr_t)p;
}
#endif

// ---------------------------------------------------------------------------
// fp32 -> f16 conversion (bandwidth-trivial prologue)
// ---------------------------------------------------------------------------
__global__ void cvt_f32_f16_kernel(const float* __restrict__ in,
                                   half_t* __restrict__ out, int n) {
    int i = blockIdx.x * blockDim.x + threadIdx.x;
    int stride = gridDim.x * blockDim.x;
    for (; i < n; i += stride) out[i] = (half_t)in[i];
}

// ---------------------------------------------------------------------------
// f16 WMMA GEMM:  C[m,n] = sum_k A[m,k] * W[n,k] + bias[n]
// Workgroup: 256 thr (8 waves), tile 64(M) x 128(N), K slab 32, double-
// buffered LDS fed by the Tensor Data Mover (wave 0 issues, TENSORcnt sync).
// Wave tile: 32x32 = 2x2 v_wmma_f32_16x16x32_f16.
//   mode 0: f16 out in [B,H,S,DK] layout;  mode 1: f32 out row-major + bias
// ---------------------------------------------------------------------------
__global__ __launch_bounds__(256)
void gemm_wmma_kernel(const half_t* __restrict__ A,
                      const half_t* __restrict__ W,
                      const float*  __restrict__ bias,
                      half_t* __restrict__ outH,
                      float*  __restrict__ outF,
                      int M, int N, int K, int mode)
{
    __shared__ __align__(16) half_t sA[2][64][32];
    __shared__ __align__(16) half_t sB[2][128][32];   // [n][k]: frag reads contiguous

    const int tid     = threadIdx.x;
    const int wave    = tid >> 5;
    const int l15     = tid & 15;
    const int halfSel = (tid >> 4) & 1;     // lane>>4

    const int m0 = blockIdx.y * 64;
    const int n0 = blockIdx.x * 128;
    const int wm = wave >> 2;               // 0..1 -> 32 rows
    const int wn = wave & 3;                // 0..3 -> 32 cols

    v8f acc[2][2];
    const v8f z8 = {};
    for (int i = 0; i < 2; ++i)
        for (int j = 0; j < 2; ++j) acc[i][j] = z8;

    const int NS = K >> 5;                  // K/32 slabs

#if HAVE_TDM
    if (wave == 0) {                        // prologue: slab 0 -> buffer 0
        tdm_load_2d(lds_off(&sA[0][0][0]), &A[(size_t)m0 * K], 32, 64,
                    (unsigned)K, (unsigned)M, (unsigned long long)K);
        tdm_load_2d(lds_off(&sB[0][0][0]), &W[(size_t)n0 * K], 32, 128,
                    (unsigned)K, (unsigned)N, (unsigned long long)K);
    }
#endif

    for (int i = 0; i < NS; ++i) {
        const int cur = i & 1;
#if HAVE_TDM
        const int nxt = cur ^ 1;
        __syncthreads();                    // readers of buf[nxt] (iter i-1) done
        if (wave == 0) {
            if (i + 1 < NS) {
                const int k1 = (i + 1) << 5;
                tdm_load_2d(lds_off(&sA[nxt][0][0]), &A[(size_t)m0 * K + k1],
                            32, 64, (unsigned)K, (unsigned)M, (unsigned long long)K);
                tdm_load_2d(lds_off(&sB[nxt][0][0]), &W[(size_t)n0 * K + k1],
                            32, 128, (unsigned)K, (unsigned)N, (unsigned long long)K);
                __builtin_amdgcn_s_wait_tensorcnt(2);   // oldest slab complete
            } else {
                __builtin_amdgcn_s_wait_tensorcnt(0);
            }
        }
        __syncthreads();                    // slab `cur` visible to all waves
#else
        const int k0 = i << 5;
        __syncthreads();
        {   // A tile 64x32 halves: 256 thr x 16B
            int row = tid >> 2, seg = tid & 3;
            float4 v = *reinterpret_cast<const float4*>(
                &A[(size_t)(m0 + row) * K + k0 + seg * 8]);
            *reinterpret_cast<float4*>(&sA[cur][row][seg * 8]) = v;
        }
        {   // B tile 128x32 halves: 256 thr x 32B
            int n = tid >> 1, seg = tid & 1;
            const half_t* src = &W[(size_t)(n0 + n) * K + k0 + seg * 16];
            float4 v0 = *reinterpret_cast<const float4*>(src);
            float4 v1 = *reinterpret_cast<const float4*>(src + 8);
            *reinterpret_cast<float4*>(&sB[cur][n][seg * 16])     = v0;
            *reinterpret_cast<float4*>(&sB[cur][n][seg * 16 + 8]) = v1;
        }
        __syncthreads();
#endif

        // A fragments: contiguous k = 16*halfSel + e (same order as B)
        v16h afrag[2];
        #pragma unroll
        for (int r = 0; r < 2; ++r) {
            int row = wm * 32 + r * 16 + l15;
            int kh  = halfSel * 16;
            #pragma unroll
            for (int e = 0; e < 8; ++e) {
                afrag[r][e]     = sA[cur][row][kh + e];
                afrag[r][8 + e] = sA[cur][row][kh + 8 + e];
            }
        }
        // B fragments: contiguous k = 16*halfSel + e (same order as A)
        v16h bfrag[2];
        #pragma unroll
        for (int c = 0; c < 2; ++c) {
            int col = wn * 32 + c * 16 + l15;
            int ko  = halfSel * 16;
            #pragma unroll
            for (int e = 0; e < 16; ++e) bfrag[c][e] = sB[cur][col][ko + e];
        }
        #pragma unroll
        for (int r = 0; r < 2; ++r)
            #pragma unroll
            for (int c = 0; c < 2; ++c)
                acc[r][c] = __builtin_amdgcn_wmma_f32_16x16x32_f16(
                    false, afrag[r], false, bfrag[c],
                    (short)0, acc[r][c], false, false);
    }

    // Epilogue. D layout: lane -> n=l15 ; VGPR r -> m = r + halfSel*8
    #pragma unroll
    for (int i = 0; i < 2; ++i) {
        int mBase = m0 + wm * 32 + i * 16 + halfSel * 8;
        #pragma unroll
        for (int j = 0; j < 2; ++j) {
            int n  = n0 + wn * 32 + j * 16 + l15;
            float bv = bias[n];
            #pragma unroll
            for (int r = 0; r < 8; ++r) {
                int m = mBase + r;
                float val = acc[i][j][r] + bv;
                if (mode == 0) {
                    int b = m >> 11;            // / S_LEN
                    int s = m & (S_LEN - 1);
                    int h = n >> 6;             // / DK
                    int d = n & (DK - 1);
                    outH[(((size_t)b * NHEAD + h) * S_LEN + s) * DK + d] = (half_t)val;
                } else {
                    outF[(size_t)m * N + n] = val;
                }
            }
        }
    }
}

// ---------------------------------------------------------------------------
// Flash attention per head. Grid: (S/64, B*H). Block: 128 thr = 4 waves,
// each wave owns 16 queries. K/V tiles (32 keys x 64 d, contiguous) are
// double-buffered in LDS via TDM; scores + P@V via WMMA; online softmax in
// f32 with shfl_xor row reductions; P re-laned through per-wave LDS tile.
// ---------------------------------------------------------------------------
__global__ __launch_bounds__(128)
void attn_wmma_kernel(const half_t* __restrict__ qP,
                      const half_t* __restrict__ kP,
                      const half_t* __restrict__ vP,
                      half_t* __restrict__ attnOut)   // [B, S, H*DK]
{
    __shared__ __align__(16) half_t sK[2][32][DK];
    __shared__ __align__(16) half_t sV[2][32][DK];
    __shared__ __align__(16) half_t sP[4][16][32];

    const int tid     = threadIdx.x;
    const int wave    = tid >> 5;
    const int lane    = tid & 31;
    const int l15     = lane & 15;
    const int halfSel = lane >> 4;

    const int bh    = blockIdx.y;
    const int qRow0 = blockIdx.x * 64 + wave * 16;

    const half_t* qHead = qP + (size_t)bh * S_LEN * DK;
    const half_t* kHead = kP + (size_t)bh * S_LEN * DK;
    const half_t* vHead = vP + (size_t)bh * S_LEN * DK;

    // Q A-fragments for the two 32-deep d-chunks (d_k = 64)
    v16h aQ[2];
    {
        const half_t* qrow = qHead + (size_t)(qRow0 + l15) * DK;
        #pragma unroll
        for (int c = 0; c < 2; ++c) {
            int kh = c * 32 + halfSel * 16;
            #pragma unroll
            for (int e = 0; e < 8; ++e) {
                aQ[c][e]     = qrow[kh + e];
                aQ[c][8 + e] = qrow[kh + 8 + e];
            }
        }
    }

    float mrun[8], lrun[8];
    #pragma unroll
    for (int j = 0; j < 8; ++j) { mrun[j] = -1e30f; lrun[j] = 0.0f; }
    v8f accO[4];
    const v8f z8 = {};
    #pragma unroll
    for (int t = 0; t < 4; ++t) accO[t] = z8;

    const float scale = 0.125f;             // 1/sqrt(64)
    const int NSLAB = S_LEN / 32;           // 64 key blocks

#if HAVE_TDM
    if (wave == 0) {                        // prologue: slab 0
        tdm_load_2d(lds_off(&sK[0][0][0]), kHead, DK, 32,
                    DK, S_LEN, (unsigned long long)DK);
        tdm_load_2d(lds_off(&sV[0][0][0]), vHead, DK, 32,
                    DK, S_LEN, (unsigned long long)DK);
    }
#endif

    for (int ib = 0; ib < NSLAB; ++ib) {
        const int cur = ib & 1;
#if HAVE_TDM
        const int nxt = cur ^ 1;
        __syncthreads();                    // buf[nxt] readers (iter ib-1) done
        if (wave == 0) {
            if (ib + 1 < NSLAB) {
                const size_t kb1 = (size_t)(ib + 1) * 32 * DK;
                tdm_load_2d(lds_off(&sK[nxt][0][0]), kHead + kb1, DK, 32,
                            DK, S_LEN, (unsigned long long)DK);
                tdm_load_2d(lds_off(&sV[nxt][0][0]), vHead + kb1, DK, 32,
                            DK, S_LEN, (unsigned long long)DK);
                __builtin_amdgcn_s_wait_tensorcnt(2);
            } else {
                __builtin_amdgcn_s_wait_tensorcnt(0);
            }
        }
        __syncthreads();
#else
        __syncthreads();
        {   // cooperative load: 2 tiles x 2048 halves, 128 thr x 16 halves each
            int row = tid >> 2, seg = tid & 3;
            const size_t src = (size_t)(ib * 32 + row) * DK + seg * 16;
            float4 k0 = *reinterpret_cast<const float4*>(&kHead[src]);
            float4 k1 = *reinterpret_cast<const float4*>(&kHead[src + 8]);
            float4 v0 = *reinterpret_cast<const float4*>(&vHead[src]);
            float4 v1 = *reinterpret_cast<const float4*>(&vHead[src + 8]);
            *reinterpret_cast<float4*>(&sK[cur][row][seg * 16])     = k0;
            *reinterpret_cast<float4*>(&sK[cur][row][seg * 16 + 8]) = k1;
            *reinterpret_cast<float4*>(&sV[cur][row][seg * 16])     = v0;
            *reinterpret_cast<float4*>(&sV[cur][row][seg * 16 + 8]) = v1;
        }
        __syncthreads();
#endif

        // ---- scores S = Q @ K^T, 16x32 as two 16x16 D fragments ----
        v8f sc[2];
        sc[0] = z8; sc[1] = z8;
        #pragma unroll
        for (int t = 0; t < 2; ++t) {
            int keyLocal = t * 16 + l15;
            #pragma unroll
            for (int c = 0; c < 2; ++c) {
                v16h bK;
                int ko = c * 32 + halfSel * 16;   // contiguous in sK row
                #pragma unroll
                for (int e = 0; e < 16; ++e) bK[e] = sK[cur][keyLocal][ko + e];
                sc[t] = __builtin_amdgcn_wmma_f32_16x16x32_f16(
                    false, aQ[c], false, bK, (short)0, sc[t], false, false);
            }
        }

        // ---- online softmax (rows live across each 16-lane half) ----
        float p0[8], p1[8], alpha[8];
        #pragma unroll
        for (int j = 0; j < 8; ++j) {
            float s0 = sc[0][j] * scale;
            float s1 = sc[1][j] * scale;
            float rmax = fmaxf(s0, s1);
            rmax = fmaxf(rmax, __shfl_xor(rmax, 1, 32));
            rmax = fmaxf(rmax, __shfl_xor(rmax, 2, 32));
            rmax = fmaxf(rmax, __shfl_xor(rmax, 4, 32));
            rmax = fmaxf(rmax, __shfl_xor(rmax, 8, 32));
            float mnew = fmaxf(mrun[j], rmax);
            float a  = __expf(mrun[j] - mnew);
            p0[j]    = __expf(s0 - mnew);
            p1[j]    = __expf(s1 - mnew);
            float rs = p0[j] + p1[j];
            rs += __shfl_xor(rs, 1, 32);
            rs += __shfl_xor(rs, 2, 32);
            rs += __shfl_xor(rs, 4, 32);
            rs += __shfl_xor(rs, 8, 32);
            lrun[j]  = lrun[j] * a + rs;
            mrun[j]  = mnew;
            alpha[j] = a;
        }
        #pragma unroll
        for (int t = 0; t < 4; ++t)
            #pragma unroll
            for (int j = 0; j < 8; ++j) accO[t][j] *= alpha[j];

        // ---- re-lane P (C/D layout -> A layout) through per-wave LDS ----
        #pragma unroll
        for (int j = 0; j < 8; ++j) {
            sP[wave][j + halfSel * 8][l15]      = (half_t)p0[j];
            sP[wave][j + halfSel * 8][16 + l15] = (half_t)p1[j];
        }
        __syncthreads();
        v16h aPf;
        {
            int kh = halfSel * 16;
            #pragma unroll
            for (int e = 0; e < 8; ++e) {
                aPf[e]     = sP[wave][l15][kh + e];
                aPf[8 + e] = sP[wave][l15][kh + 8 + e];
            }
        }

        // ---- out += P @ V (4 d-tiles of 16) ----
        #pragma unroll
        for (int t = 0; t < 4; ++t) {
            v16h bV;
            int d    = t * 16 + l15;
            int kOff = halfSel * 16;
            #pragma unroll
            for (int e = 0; e < 16; ++e) bV[e] = sV[cur][kOff + e][d];
            accO[t] = __builtin_amdgcn_wmma_f32_16x16x32_f16(
                false, aPf, false, bV, (short)0, accO[t], false, false);
        }
    }

    // ---- normalize + write merged-head layout [B, S, H*DK] as f16 ----
    int b = bh >> 4;
    int h = bh & (NHEAD - 1);
    #pragma unroll
    for (int t = 0; t < 4; ++t) {
        int d = t * 16 + l15;
        #pragma unroll
        for (int j = 0; j < 8; ++j) {
            int m = j + halfSel * 8;
            int s = qRow0 + m;
            float inv = 1.0f / lrun[j];
            attnOut[(((size_t)b * S_LEN + s) * NHEAD + h) * DK + d] =
                (half_t)(accO[t][j] * inv);
        }
    }
}

// ---------------------------------------------------------------------------
// Host-side orchestration
// ---------------------------------------------------------------------------
extern "C" void kernel_launch(void* const* d_in, const int* in_sizes, int n_in,
                              void* d_out, int out_size, void* d_ws, size_t ws_size,
                              hipStream_t stream) {
    const float* Q   = (const float*)d_in[0];
    const float* K   = (const float*)d_in[1];
    const float* V   = (const float*)d_in[2];
    const float* W_q = (const float*)d_in[3];
    const float* b_q = (const float*)d_in[4];
    const float* W_k = (const float*)d_in[5];
    const float* b_k = (const float*)d_in[6];
    const float* W_v = (const float*)d_in[7];
    const float* b_v = (const float*)d_in[8];
    const float* W_o = (const float*)d_in[9];
    const float* b_o = (const float*)d_in[10];
    float* out = (float*)d_out;

    const size_t ACT = (size_t)MROWS * DMODEL;     // 4M elements
    const size_t WEL = (size_t)DMODEL * DMODEL;    // 1M elements

    half_t* ws   = (half_t*)d_ws;
    half_t* Qh   = ws;                 // f16 activations
    half_t* Kh   = Qh  + ACT;
    half_t* Vh   = Kh  + ACT;
    half_t* Wqh  = Vh  + ACT;          // f16 weights
    half_t* Wkh  = Wqh + WEL;
    half_t* Wvh  = Wkh + WEL;
    half_t* Woh  = Wvh + WEL;
    half_t* qPh  = Woh + WEL;          // projections, [B,H,S,DK]
    half_t* kPh  = qPh + ACT;
    half_t* vPh  = kPh + ACT;
    half_t* attO = vPh + ACT;          // attention out, [B,S,H*DK]

    const int CB = 256, CG = 1024;
    cvt_f32_f16_kernel<<<CG, CB, 0, stream>>>(Q,   Qh,  (int)ACT);
    cvt_f32_f16_kernel<<<CG, CB, 0, stream>>>(K,   Kh,  (int)ACT);
    cvt_f32_f16_kernel<<<CG, CB, 0, stream>>>(V,   Vh,  (int)ACT);
    cvt_f32_f16_kernel<<<CG, CB, 0, stream>>>(W_q, Wqh, (int)WEL);
    cvt_f32_f16_kernel<<<CG, CB, 0, stream>>>(W_k, Wkh, (int)WEL);
    cvt_f32_f16_kernel<<<CG, CB, 0, stream>>>(W_v, Wvh, (int)WEL);
    cvt_f32_f16_kernel<<<CG, CB, 0, stream>>>(W_o, Woh, (int)WEL);

    dim3 gP(DMODEL / 128, MROWS / 64);  // (8, 64)
    gemm_wmma_kernel<<<gP, 256, 0, stream>>>(Qh, Wqh, b_q, qPh, nullptr,
                                             MROWS, DMODEL, DMODEL, 0);
    gemm_wmma_kernel<<<gP, 256, 0, stream>>>(Kh, Wkh, b_k, kPh, nullptr,
                                             MROWS, DMODEL, DMODEL, 0);
    gemm_wmma_kernel<<<gP, 256, 0, stream>>>(Vh, Wvh, b_v, vPh, nullptr,
                                             MROWS, DMODEL, DMODEL, 0);

    dim3 gA(S_LEN / 64, BATCH * NHEAD); // (32, 32)
    attn_wmma_kernel<<<gA, 128, 0, stream>>>(qPh, kPh, vPh, attO);

    gemm_wmma_kernel<<<gP, 256, 0, stream>>>(attO, Woh, b_o, nullptr, out,
                                             MROWS, DMODEL, DMODEL, 1);
}
